// MSACrossAttention_48610439856674
// MI455X (gfx1250) — matrix operand order repaired and checked by reading, not compiled, not run on hardware
//
#include <hip/hip_runtime.h>
#include <hip/hip_bf16.h>
#include <stdint.h>

// ---------------------------------------------------------------------------
// MSA cross attention pipeline for gfx1250 (MI455X), wave32.
//   b=2 m=8 s=4096 d=1024 H=16 C=4 da=256, N = b*s = 8192
//   1) convert x, w_q, w_kv, w_o   fp32 -> bf16
//   2) kv  = x2 @ w_kv^T + b_kv    (65536x1024)x(1024x2048)  WMMA bf16
//   3) Q   = x0 @ w_q^T  + b_q     (8192x1024)x(1024x4096)   WMMA bf16
//   4) attention per (n,c)         VALU fp32
//   5) ypre = o @ w_o^T + b_o      (8192x4096)x(4096x1024)   WMMA bf16
//   6) y = ypre + x0; RMSNorm -> out (fp32)
// GEMM staging now uses the Tensor Data Mover (TENSOR_LOAD_TO_LDS) with
// hardware LDS padding + TENSORcnt double buffering when available.
// ---------------------------------------------------------------------------

typedef __attribute__((ext_vector_type(16))) __bf16 v16bf;
typedef __attribute__((ext_vector_type(8)))  float  v8f;
typedef __attribute__((ext_vector_type(4)))  unsigned int u32x4;
typedef __attribute__((ext_vector_type(8)))  int i32x8;
typedef __attribute__((ext_vector_type(4)))  int i32x4;

#if __has_builtin(__builtin_amdgcn_tensor_load_to_lds)
#define USE_TDM 1
#else
#define USE_TDM 0
#endif

union FragU {
    uint4 q[2];
    v16bf v;
};

__device__ __forceinline__ unsigned short f32_to_bf16_rne(float f) {
    union { float f; unsigned int u; } cv;
    cv.f = f;
    unsigned int u = cv.u;
    unsigned int r = u + 0x7FFFu + ((u >> 16) & 1u);
    return (unsigned short)(r >> 16);
}

// ---------------------------------------------------------------------------
__global__ __launch_bounds__(256) void f32_to_bf16_kernel(
    const float* __restrict__ src, unsigned short* __restrict__ dst, long long n)
{
    long long i = (long long)blockIdx.x * blockDim.x + threadIdx.x;
    long long stride = (long long)gridDim.x * blockDim.x;
    for (; i < n; i += stride)
        dst[i] = f32_to_bf16_rne(src[i]);
}

// ---------------------------------------------------------------------------
// TDM: 2D tile (rows x 32 bf16) -> LDS with padded rows (40 ushorts = 80 B).
// D# group0: count=1, lds_addr, global_addr, type=2.
// D# group1: data_size=2B, pad_enable, pad_interval code 3 (16 DWORD = 64 B,
//            i.e. one 32-bf16 row), pad_amount code 3 (4 DWORD = 16 B),
//            tensor_dim0 = tile_dim0 = 32, tensor_dim1 = tile_dim1 = rows,
//            tensor_dim0_stride = stride_elems.
// ---------------------------------------------------------------------------
#if USE_TDM
__device__ __forceinline__ void tdm_load_tile_2d(
    unsigned lds_byte_off, const void* gaddr,
    unsigned rows, unsigned stride_elems)
{
    unsigned long long ga = (unsigned long long)(uintptr_t)gaddr;
    u32x4 g0;
    g0[0] = 1u;                                        // count=1 (valid), user D#
    g0[1] = lds_byte_off;                              // bits [63:32] lds_addr
    g0[2] = (unsigned)(ga & 0xFFFFFFFFull);            // global_addr[31:0]
    g0[3] = (unsigned)((ga >> 32) & 0x01FFFFFFull)     // global_addr[56:32]
            | (2u << 30);                              // type = 2 ("image")
    i32x8 g1;
    g1[0] = (int)((1u << 16)      // data_size = 2 bytes
                | (1u << 20)      // pad_enable
                | (3u << 22)      // pad_interval: 16 DWORDs (one 64B row)
                | (3u << 25));    // pad_amount:   4 DWORDs (16B)
    g1[1] = (int)(32u << 16);                 // tensor_dim0[15:0] = 32
    g1[2] = (int)((rows & 0xFFFFu) << 16);    // tensor_dim1[15:0] = rows
    g1[3] = (int)(32u << 16);                 // tile_dim0 = 32
    g1[4] = (int)(rows & 0xFFFFu);            // tile_dim1 = rows, tile_dim2 = 0
    g1[5] = (int)stride_elems;                // tensor_dim0_stride[31:0]
    g1[6] = 0;
    g1[7] = 0;
    i32x4 z4 = {0, 0, 0, 0};
#if defined(__clang_major__) && (__clang_major__ >= 23)
    i32x8 z8 = {0, 0, 0, 0, 0, 0, 0, 0};
    __builtin_amdgcn_tensor_load_to_lds(g0, g1, z4, z4, z8, 0);
#else
    __builtin_amdgcn_tensor_load_to_lds(g0, g1, z4, z4, 0);
#endif
}
#endif

// ---------------------------------------------------------------------------
// Generic tiled GEMM: C[M,N] (f32) = A[M,K](bf16, row stride lda) * B[N,K]^T
//                                    + bias[N]
// Block 256 threads (8 waves), tile 128(M) x 64(N), K-step 32.
// Each wave computes 32x32 = 2x2 WMMA 16x16 tiles.
// Fragment layouts per CDNA5 ISA (see Round-1 comments).
// Requires M%128==0, N%64==0, K%32==0.
// ---------------------------------------------------------------------------
#define TILE_M 128
#define TILE_N 64
#define TILE_K 32
#define LDSW   40   // ushorts per LDS row (80 B)

__device__ __forceinline__ void wmma_step(
    const unsigned short* __restrict__ As, const unsigned short* __restrict__ Bs,
    int aRowL, int aKb, int bColL, int bKb, v8f (&acc)[2][2])
{
    FragU afrag[2], bfrag[2];
    for (int ms = 0; ms < 2; ++ms) {
        const unsigned short* p = &As[(aRowL + ms * 16) * LDSW];
        afrag[ms].q[0] = *(const uint4*)(p + aKb);        // K = kb .. kb+7
        afrag[ms].q[1] = *(const uint4*)(p + aKb + 16);   // K = kb+16 .. kb+23
    }
    for (int ns = 0; ns < 2; ++ns) {
        const unsigned short* p = &Bs[(bColL + ns * 16) * LDSW];
        bfrag[ns].q[0] = *(const uint4*)(p + bKb);        // K = kb .. kb+7
        bfrag[ns].q[1] = *(const uint4*)(p + bKb + 16);   // K = kb+16 .. kb+23
    }
    for (int ms = 0; ms < 2; ++ms)
        for (int ns = 0; ns < 2; ++ns)
            acc[ms][ns] = __builtin_amdgcn_wmma_f32_16x16x32_bf16(
                false, afrag[ms].v, false, bfrag[ns].v,
                (short)0, acc[ms][ns], false, false);
}

__global__ __launch_bounds__(256) void gemm_bf16_wmma(
    const unsigned short* __restrict__ A, long long lda,
    const unsigned short* __restrict__ B,   // N rows x K cols (weight, row-major)
    const float* __restrict__ bias,
    float* __restrict__ C,
    int M, int N, int K)
{
    const int t     = threadIdx.x;
    const int lane  = t & 31;
    const int wave  = t >> 5;
    const int waveM = wave >> 1;   // 0..3 -> M offset *32
    const int waveN = wave & 1;    // 0..1 -> N offset *32
    const int blockN0 = blockIdx.x * TILE_N;
    const int blockM0 = blockIdx.y * TILE_M;

    v8f acc[2][2];
    for (int a = 0; a < 2; ++a)
        for (int b = 0; b < 2; ++b)
            for (int v = 0; v < 8; ++v)
                acc[a][b][v] = 0.0f;

    // per-lane fragment addressing (constant across K loop)
    const int aRowL = waveM * 32 + (lane & 15);
    const int aKb   = (lane >> 4) * 8;        // 0 or 8
    const int bColL = waveN * 32 + (lane & 15);
    const int bKb   = (lane >> 4) * 8;        // 0 or 8

#if USE_TDM
    // ---- Tensor Data Mover double-buffered staging --------------------
    __shared__ alignas(16) unsigned short As[2][TILE_M * LDSW];
    __shared__ alignas(16) unsigned short Bs[2][TILE_N * LDSW];
    const bool tdmWave = (t < 32);   // one wave issues the DMA for the block

    const unsigned short* aTileBase = A + (long long)blockM0 * lda;
    const unsigned short* bTileBase = B + (long long)blockN0 * K;

    if (tdmWave) {
        tdm_load_tile_2d((unsigned)(uintptr_t)&As[0][0], aTileBase, TILE_M, (unsigned)lda);
        tdm_load_tile_2d((unsigned)(uintptr_t)&Bs[0][0], bTileBase, TILE_N, (unsigned)K);
    }
    int buf = 0;
    for (int k0 = 0; k0 < K; k0 += TILE_K) {
        if (tdmWave) {
            if (k0 + TILE_K < K) {
                // prefetch next tile pair into the other buffer
                tdm_load_tile_2d((unsigned)(uintptr_t)&As[buf ^ 1][0],
                                 aTileBase + (k0 + TILE_K), TILE_M, (unsigned)lda);
                tdm_load_tile_2d((unsigned)(uintptr_t)&Bs[buf ^ 1][0],
                                 bTileBase + (k0 + TILE_K), TILE_N, (unsigned)K);
                __builtin_amdgcn_s_wait_tensorcnt(2);  // current pair landed
            } else {
                __builtin_amdgcn_s_wait_tensorcnt(0);
            }
        }
        __syncthreads();   // current buffer visible to all waves
        wmma_step(&As[buf][0], &Bs[buf][0], aRowL, aKb, bColL, bKb, acc);
        __syncthreads();   // all reads done before TDM overwrites this buffer
        buf ^= 1;
    }
#else
    // ---- synchronous fallback staging ---------------------------------
    __shared__ alignas(16) unsigned short As[TILE_M * LDSW];
    __shared__ alignas(16) unsigned short Bs[TILE_N * LDSW];

    const int arow  = t >> 1;                 // 0..127
    const int ahalf = t & 1;                  // 16 bf16 per chunk
    const unsigned short* aSrc = A + (long long)(blockM0 + arow) * lda + ahalf * 16;
    unsigned short* aDst = &As[arow * LDSW + ahalf * 16];
    const int brow  = (t & 127) >> 1;         // 0..63
    const int bhalf = t & 1;
    const unsigned short* bSrc = B + (long long)(blockN0 + brow) * K + bhalf * 16;
    unsigned short* bDst = &Bs[brow * LDSW + bhalf * 16];
    const bool doB = t < 128;

    for (int k0 = 0; k0 < K; k0 += TILE_K) {
        *(uint4*)aDst = *(const uint4*)(aSrc + k0);
        if (doB)
            *(uint4*)bDst = *(const uint4*)(bSrc + k0);
        if (k0 + TILE_K < K) {
            __builtin_prefetch(aSrc + k0 + TILE_K, 0, 0);
            __builtin_prefetch(bSrc + k0 + TILE_K, 0, 0);
        }
        __syncthreads();
        wmma_step(As, Bs, aRowL, aKb, bColL, bKb, acc);
        __syncthreads();
    }
#endif

    // store D + bias
    const int lhalf = lane >> 4;   // 0: M 0-7, 1: M 8-15
    for (int ms = 0; ms < 2; ++ms) {
        const int rbase = blockM0 + waveM * 32 + ms * 16 + lhalf * 8;
        for (int ns = 0; ns < 2; ++ns) {
            const int col = blockN0 + waveN * 32 + ns * 16 + (lane & 15);
            const float bv = bias[col];
            for (int v = 0; v < 8; ++v)
                C[(long long)(rbase + v) * N + col] = acc[ms][ns][v] + bv;
        }
    }
}

// ---------------------------------------------------------------------------
// Attention: one wave32 per (n, c). Lane l covers dims kd = l + 32*j, j=0..7.
// K[n,c,mi,kd] = kv[(n*8 + 2c + mi/4)*2048 + (mi&3)*512 + kd]; V same + 256.
// scale = 1/sqrt(256) = 1/16. Writes o (bf16) at oh[n*4096 + c*1024 + g*256+kd].
// ---------------------------------------------------------------------------
__global__ __launch_bounds__(256) void attn_kernel(
    const float* __restrict__ Qf, const float* __restrict__ kvf,
    unsigned short* __restrict__ oh)
{
    const int wave = threadIdx.x >> 5;
    const int lane = threadIdx.x & 31;
    const int p = blockIdx.x * 8 + wave;   // (n,c) pair index, < 8192*4
    const int n = p >> 2;
    const int c = p & 3;

    const float* qbase = Qf + (long long)n * 4096 + c * 1024;
    float q[4][8];
    for (int g = 0; g < 4; ++g)
        for (int j = 0; j < 8; ++j)
            q[g][j] = qbase[g * 256 + lane + 32 * j];

    float s[4][8];
    for (int mi = 0; mi < 8; ++mi) {
        const float* kbase = kvf + ((long long)n * 8 + 2 * c + (mi >> 2)) * 2048
                                 + (mi & 3) * 512;
        float kr[8];
        for (int j = 0; j < 8; ++j) kr[j] = kbase[lane + 32 * j];
        for (int g = 0; g < 4; ++g) {
            float part = 0.0f;
            for (int j = 0; j < 8; ++j) part += q[g][j] * kr[j];
            for (int off = 16; off > 0; off >>= 1)
                part += __shfl_xor(part, off, 32);
            s[g][mi] = part * 0.0625f;
        }
    }

    float attn[4][8];
    for (int g = 0; g < 4; ++g) {
        float mx = s[g][0];
        for (int mi = 1; mi < 8; ++mi) mx = fmaxf(mx, s[g][mi]);
        float den = 0.0f;
        for (int mi = 0; mi < 8; ++mi) { attn[g][mi] = expf(s[g][mi] - mx); den += attn[g][mi]; }
        float inv = 1.0f / den;
        for (int mi = 0; mi < 8; ++mi) attn[g][mi] *= inv;
    }

    float o[4][8];
    for (int g = 0; g < 4; ++g)
        for (int j = 0; j < 8; ++j) o[g][j] = 0.0f;
    for (int mi = 0; mi < 8; ++mi) {
        const float* vbase = kvf + ((long long)n * 8 + 2 * c + (mi >> 2)) * 2048
                                 + (mi & 3) * 512 + 256;
        float vr[8];
        for (int j = 0; j < 8; ++j) vr[j] = vbase[lane + 32 * j];
        for (int g = 0; g < 4; ++g)
            for (int j = 0; j < 8; ++j)
                o[g][j] += attn[g][mi] * vr[j];
    }

    unsigned short* obase = oh + (long long)n * 4096 + c * 1024;
    for (int g = 0; g < 4; ++g)
        for (int j = 0; j < 8; ++j)
            obase[g * 256 + lane + 32 * j] = f32_to_bf16_rne(o[g][j]);
}

// ---------------------------------------------------------------------------
// y = ypre + x0;  out = y / sqrt(mean(y^2) + eps) * norm_w.  One block per row.
// ---------------------------------------------------------------------------
__global__ __launch_bounds__(256) void resnorm_kernel(
    const float* __restrict__ ypre, const float* __restrict__ x,
    const float* __restrict__ norm_w, float* __restrict__ out)
{
    __shared__ float red[256];
    const int n = blockIdx.x;
    const int t = threadIdx.x;
    const float* yb = ypre + (long long)n * 1024;
    const float* xb = x + (long long)n * 8 * 1024;

    float y[4];
    float ss = 0.0f;
    for (int j = 0; j < 4; ++j) {
        const int i = t + 256 * j;
        y[j] = yb[i] + xb[i];
        ss += y[j] * y[j];
    }
    red[t] = ss;
    __syncthreads();
    for (int off = 128; off > 0; off >>= 1) {
        if (t < off) red[t] += red[t + off];
        __syncthreads();
    }
    const float rinv = 1.0f / sqrtf(red[0] * (1.0f / 1024.0f) + 1e-6f);
    for (int j = 0; j < 4; ++j) {
        const int i = t + 256 * j;
        out[(long long)n * 1024 + i] = y[j] * rinv * norm_w[i];
    }
}

// ---------------------------------------------------------------------------
extern "C" void kernel_launch(void* const* d_in, const int* in_sizes, int n_in,
                              void* d_out, int out_size, void* d_ws, size_t ws_size,
                              hipStream_t stream) {
    const float* x      = (const float*)d_in[0];   // (2,8,4096,1024)
    const float* w_q    = (const float*)d_in[1];   // (4096,1024)
    const float* b_q    = (const float*)d_in[2];   // (4096,)
    const float* w_kv   = (const float*)d_in[3];   // (2048,1024)
    const float* b_kv   = (const float*)d_in[4];   // (2048,)
    const float* w_o    = (const float*)d_in[5];   // (1024,4096)
    const float* b_o    = (const float*)d_in[6];   // (1024,)
    const float* norm_w = (const float*)d_in[7];   // (1024,)
    float* out = (float*)d_out;                    // (2,4096,1024)

    const long long NX   = 2LL * 8 * 4096 * 1024;      // 67,108,864 x elements
    const long long NROW = 8192;                       // b*s
    const long long NKVR = NROW * 8;                   // 65536 kv rows

    char* ws = (char*)d_ws;
    size_t off = 0;
    unsigned short* xh   = (unsigned short*)(ws + off); off += (size_t)NX * 2;
    unsigned short* wqh  = (unsigned short*)(ws + off); off += (size_t)4096 * 1024 * 2;
    unsigned short* wkvh = (unsigned short*)(ws + off); off += (size_t)2048 * 1024 * 2;
    unsigned short* woh  = (unsigned short*)(ws + off); off += (size_t)1024 * 4096 * 2;
    float*          kvf  = (float*)(ws + off);          off += (size_t)NKVR * 2048 * 4;
    float*          Qf   = (float*)(ws + off);          off += (size_t)NROW * 4096 * 4;
    unsigned short* ohb  = (unsigned short*)(ws + off); off += (size_t)NROW * 4096 * 2;
    float*          ypre = (float*)(ws + off);          off += (size_t)NROW * 1024 * 4;

    // 1) convert activations + weights to bf16
    f32_to_bf16_kernel<<<4096, 256, 0, stream>>>(x,    xh,   NX);
    f32_to_bf16_kernel<<<1024, 256, 0, stream>>>(w_q,  wqh,  4096LL * 1024);
    f32_to_bf16_kernel<<<512,  256, 0, stream>>>(w_kv, wkvh, 2048LL * 1024);
    f32_to_bf16_kernel<<<1024, 256, 0, stream>>>(w_o,  woh,  1024LL * 4096);

    // 2) kv = x2_flat @ w_kv^T + b_kv : M=65536 N=2048 K=1024, lda=1024
    gemm_bf16_wmma<<<dim3(2048 / TILE_N, 65536 / TILE_M), 256, 0, stream>>>(
        xh, 1024, wkvh, b_kv, kvf, 65536, 2048, 1024);

    // 3) Q = x0 @ w_q^T + b_q : M=8192 N=4096 K=1024, lda=8*1024 (every 8th row)
    gemm_bf16_wmma<<<dim3(4096 / TILE_N, 8192 / TILE_M), 256, 0, stream>>>(
        xh, 8192, wqh, b_q, Qf, 8192, 4096, 1024);

    // 4) attention -> o (bf16): 8192*4 (n,c) pairs, one wave each
    attn_kernel<<<4096, 256, 0, stream>>>(Qf, kvf, ohb);

    // 5) ypre = o @ w_o^T + b_o : M=8192 N=1024 K=4096, lda=4096
    gemm_bf16_wmma<<<dim3(1024 / TILE_N, 8192 / TILE_M), 256, 0, stream>>>(
        ohb, 4096, woh, b_o, ypre, 8192, 1024, 4096);

    // 6) residual + RMSNorm
    resnorm_kernel<<<8192, 256, 0, stream>>>(ypre, x, norm_w, out);
}
